// Conv2D_10634339025228
// MI455X (gfx1250) — matrix operand order repaired and checked by reading, not compiled, not run on hardware
//
#include <hip/hip_runtime.h>

typedef __bf16 bf16_t;
typedef bf16_t v16bf __attribute__((ext_vector_type(16)));
typedef float  v8f   __attribute__((ext_vector_type(8)));

#define CIN   128
#define HDIM  56
#define WDIM  56
#define COUT  256
#define NIMG  32
#define HWSZ  (HDIM * WDIM)          // 3136
#define KDIM  (CIN * 9)              // 1152
#define NPIX  (NIMG * HWSZ)          // 100352
#define BM    128                    // C_out tile per block
#define BN    64                     // pixel tile per block

// ---- main kernel tiling (BK=64: two WMMA K-halves per step) ----
#define BKM    64
#define KSTEPSM (KDIM / BKM)         // 18
#define LDTM   72                    // padded LDS row (bf16): 64 + 8
#define SA_OFF(b)  ((b) * BM * LDTM)                   // 0 / 9216 (ushorts)
#define SB_OFF(b)  (2 * BM * LDTM + (b) * BN * LDTM)   // 18432 / 23040
#define SMEM_USH   (2 * BM * LDTM + 2 * BN * LDTM)     // 27648 ush = 55296 B

// ---- fallback kernel tiling ----
#define BKF    32
#define KSTEPSF (KDIM / BKF)         // 36
#define LDTF   40

// pack two f32 -> packed bf16 pair: round-half-up add + one v_perm_b32
__device__ __forceinline__ unsigned pack_bf16(float a, float b) {
  unsigned ua = __float_as_uint(a) + 0x8000u;
  unsigned ub = __float_as_uint(b) + 0x8000u;
  return __builtin_amdgcn_perm(ub, ua, 0x07060302u);  // {ub[31:16], ua[31:16]}
}

// Build a 16xbf16 fragment from two 16B LDS chunks (2x ds_load_b128).
__device__ __forceinline__ v16bf load_frag(const unsigned short* p0,
                                           const unsigned short* p1) {
  struct U { uint4 a, b; } u;
  u.a = *reinterpret_cast<const uint4*>(p0);
  u.b = *reinterpret_cast<const uint4*>(p1);
  return __builtin_bit_cast(v16bf, u);
}

// ---------------- prepass: X f32 -> bf16 (same NCHW layout) ----------------
__global__ __launch_bounds__(256) void cvt_x_bf16(const float* __restrict__ X,
                                                  unsigned short* __restrict__ Xb) {
  const int i = (blockIdx.x * 256 + threadIdx.x) * 4;
  const float4 f = *reinterpret_cast<const float4*>(X + i);
  *reinterpret_cast<uint2*>(Xb + i) =
      make_uint2(pack_bf16(f.x, f.y), pack_bf16(f.z, f.w));
}

// ---- prepass: W f32 [co][ci][kh][kw] -> bf16 [co][(kh*3+kw)*128 + ci] -----
__global__ __launch_bounds__(256) void cvt_w_bf16(const float* __restrict__ Wt,
                                                  unsigned short* __restrict__ Wb) {
  const int idx = blockIdx.x * 256 + threadIdx.x;   // destination index
  const int co = idx / KDIM;
  const int t  = idx - co * KDIM;
  const int r9 = t >> 7;                            // kh*3+kw
  const int ci = t & 127;
  const unsigned u = __float_as_uint(Wt[co * KDIM + ci * 9 + r9]) + 0x8000u;
  Wb[idx] = (unsigned short)(u >> 16);
}

// --------------------------- main WMMA kernel ------------------------------
__global__ __launch_bounds__(256) void conv3x3_wmma_bf16(
    const unsigned short* __restrict__ Xb,   // bf16 [32,128,56,56]
    const unsigned short* __restrict__ Wb,   // bf16 [256, 1152] k-reordered
    const float* __restrict__ Bias,
    float* __restrict__ Out)
{
  __shared__ unsigned short smem[SMEM_USH];

  const int tid = threadIdx.x;
  const int p0  = blockIdx.x * BN;
  const int co0 = blockIdx.y * BM;

  // ---- A tile: async global->LDS copy, 4 x b128 chunks per thread --------
  const int acr = tid >> 1;                // A row (cout) 0..127
  const int acp = (tid & 1) * 4;           // chunks 0..3 or 4..7 within row
  const unsigned short* wsrc = Wb + (size_t)(co0 + acr) * KDIM + acp * 8;
  const unsigned albase = (unsigned)(acr * (LDTM * 2) + acp * 16);

  // ---- B tile: im2col gather, 16 bf16 per thread -------------------------
  const int brow = tid >> 2;               // pixel within tile 0..63
  const int bkc  = (tid & 3) * 16;         // ci sub-offset 0,16,32,48
  const int bp  = p0 + brow;
  const int bn  = bp / HWSZ;
  const int bhw = bp - bn * HWSZ;
  const int bh  = bhw / WDIM;
  const int bw  = bhw - bh * WDIM;
  const unsigned short* xb_n = Xb + (size_t)bn * CIN * HWSZ + bkc * HWSZ;

  unsigned xg[16];
  const unsigned short* sptr = xb_n;       // gather base (updated per step)
  bool validb = false;

  auto a_async = [&](int kt, int buf) {
    const unsigned lds0 = (unsigned)(SA_OFF(buf) * 2) + albase;
    const unsigned long long g =
        (unsigned long long)(uintptr_t)(wsrc + kt * BKM);
#pragma unroll
    for (int c = 0; c < 4; ++c) {
      asm volatile("global_load_async_to_lds_b128 %0, %1, off"
                   :: "v"(lds0 + (unsigned)(c * 16)), "v"(g + (unsigned long long)(c * 16))
                   : "memory");
    }
  };

  auto b_gather = [&](int kt) {
    if ((kt & 1) == 0) {                   // uniform branch: r9 changed
      const int r9  = kt >> 1;             // kh*3+kw
      const int kh  = r9 / 3;
      const int kw  = r9 - kh * 3;
      const int hin = bh + kh - 1;
      const int win = bw + kw - 1;
      validb = ((unsigned)hin < (unsigned)HDIM) &&
               ((unsigned)win < (unsigned)WDIM);
      const int hc = min(max(hin, 0), HDIM - 1);   // clamp -> always loadable
      const int wc = min(max(win, 0), WDIM - 1);
      sptr = xb_n + hc * WDIM + wc;
    } else {
      sptr += BKM * HWSZ;                  // ci0 advances by 64
    }
#pragma unroll
    for (int e = 0; e < 16; ++e)           // unconditional, imm offsets
      xg[e] = (unsigned)sptr[e * HWSZ];
  };

  auto b_commit = [&](int buf) {
    unsigned u[8];
#pragma unroll
    for (int i = 0; i < 8; ++i) {
      unsigned p = xg[2 * i] | (xg[2 * i + 1] << 16);
      u[i] = validb ? p : 0u;              // v_cndmask, no branches
    }
    uint4* d = reinterpret_cast<uint4*>(&smem[SB_OFF(buf) + brow * LDTM + bkc]);
    d[0] = make_uint4(u[0], u[1], u[2], u[3]);
    d[1] = make_uint4(u[4], u[5], u[6], u[7]);
  };

  auto wait_async = [] { asm volatile("s_wait_asynccnt 0" ::: "memory"); };

  // ---- per-wave compute mapping ------------------------------------------
  const int lane = tid & 31;
  const int wv   = tid >> 5;
  const int msub = (wv & 3) * 32;
  const int nsub = (wv >> 2) * 32;
  const int lo   = lane & 15;
  const int hi   = lane >> 4;

  v8f accum[2][2];
#pragma unroll
  for (int tm = 0; tm < 2; ++tm)
#pragma unroll
    for (int tn = 0; tn < 2; ++tn)
      accum[tm][tn] = v8f{0.f, 0.f, 0.f, 0.f, 0.f, 0.f, 0.f, 0.f};

  a_async(0, 0);
  b_gather(0);
  b_commit(0);
  wait_async();
  __syncthreads();

  for (int kt = 0; kt < KSTEPSM; ++kt) {
    const int buf = kt & 1;
    const int nb  = buf ^ 1;
    if (kt + 1 < KSTEPSM) {
      a_async(kt + 1, nb);        // async engine fills next A tile
      b_gather(kt + 1);           // global u16 gathers overlap WMMA
    }

    const unsigned short* A  = &smem[SA_OFF(buf)];
    const unsigned short* Bs = &smem[SB_OFF(buf)];
#pragma unroll
    for (int h = 0; h < 2; ++h) {          // two K=32 halves of the BK=64 step
      v16bf bfv[2];
#pragma unroll
      for (int tn = 0; tn < 2; ++tn) {
        const unsigned short* bpp =
            Bs + (nsub + tn * 16 + lo) * LDTM + h * 32 + hi * 8;
        bfv[tn] = load_frag(bpp, bpp + 16);
      }
#pragma unroll
      for (int tm = 0; tm < 2; ++tm) {
        const unsigned short* ap = A + (msub + tm * 16 + lo) * LDTM + h * 32;
        const v16bf af = load_frag(ap + hi * 8, ap + 16 + hi * 8);
#pragma unroll
        for (int tn = 0; tn < 2; ++tn) {
          accum[tm][tn] = __builtin_amdgcn_wmma_f32_16x16x32_bf16(
              false, af, false, bfv[tn], (short)0, accum[tm][tn], false, false);
        }
      }
    }

    if (kt + 1 < KSTEPSM) {
      b_commit(nb);
      wait_async();
      __syncthreads();
    }
  }

  // ---- epilogue: bias + fp32 scatter to NCHW -----------------------------
#pragma unroll
  for (int tm = 0; tm < 2; ++tm) {
#pragma unroll
    for (int tn = 0; tn < 2; ++tn) {
      const int p  = p0 + nsub + tn * 16 + lo;
      const int n  = p / HWSZ;
      const int hw = p - n * HWSZ;
#pragma unroll
      for (int r = 0; r < 8; ++r) {
        const int co = co0 + msub + tm * 16 + r + hi * 8;  // D: M = r + 8*hi
        Out[((size_t)n * COUT + co) * HWSZ + hw] = accum[tm][tn][r] + Bias[co];
      }
    }
  }
}

// ---------------- fallback (ws too small): convert in-kernel ---------------
__global__ __launch_bounds__(256) void conv3x3_fallback(
    const float* __restrict__ X, const float* __restrict__ Wt,
    const float* __restrict__ Bias, float* __restrict__ Out)
{
  __shared__ unsigned short sA[2][BM * LDTF];
  __shared__ unsigned short sB[2][BN * LDTF];

  const int tid = threadIdx.x;
  const int p0  = blockIdx.x * BN;
  const int co0 = blockIdx.y * BM;

  const int arow = tid >> 1;
  const int akc  = (tid & 1) * 16;
  const int brow = tid >> 2;
  const int bkc  = (tid & 3) * 8;

  const int bp  = p0 + brow;
  const int bn  = bp / HWSZ;
  const int bhw = bp - bn * HWSZ;
  const int bh  = bhw / WDIM;
  const int bw  = bhw - bh * WDIM;
  const float* xbase = X + (size_t)bn * CIN * HWSZ;
  const float* wbase = Wt + (size_t)(co0 + arow) * KDIM + akc;

  float4 wreg[4];
  float  xreg[8];

  auto prefetch = [&](int kt) {
    const float4* wp = reinterpret_cast<const float4*>(wbase + kt * BKF);
    wreg[0] = wp[0]; wreg[1] = wp[1]; wreg[2] = wp[2]; wreg[3] = wp[3];
#pragma unroll
    for (int e = 0; e < 8; ++e) {
      const int k  = kt * BKF + bkc + e;
      const int ci = k / 9;
      const int r9 = k - ci * 9;
      const int kh = r9 / 3;
      const int kw = r9 - kh * 3;
      const int hin = bh + kh - 1;
      const int win = bw + kw - 1;
      float v = 0.0f;
      if ((unsigned)hin < (unsigned)HDIM && (unsigned)win < (unsigned)WDIM)
        v = xbase[(ci * HDIM + hin) * WDIM + win];
      xreg[e] = v;
    }
  };

  auto commit = [&](int buf) {
    const float* wf = reinterpret_cast<const float*>(wreg);
    uint4* da = reinterpret_cast<uint4*>(&sA[buf][arow * LDTF + akc]);
    da[0] = make_uint4(pack_bf16(wf[0], wf[1]), pack_bf16(wf[2], wf[3]),
                       pack_bf16(wf[4], wf[5]), pack_bf16(wf[6], wf[7]));
    da[1] = make_uint4(pack_bf16(wf[8], wf[9]), pack_bf16(wf[10], wf[11]),
                       pack_bf16(wf[12], wf[13]), pack_bf16(wf[14], wf[15]));
    *reinterpret_cast<uint4*>(&sB[buf][brow * LDTF + bkc]) =
        make_uint4(pack_bf16(xreg[0], xreg[1]), pack_bf16(xreg[2], xreg[3]),
                   pack_bf16(xreg[4], xreg[5]), pack_bf16(xreg[6], xreg[7]));
  };

  const int lane = tid & 31;
  const int wv   = tid >> 5;
  const int msub = (wv & 3) * 32;
  const int nsub = (wv >> 2) * 32;
  const int lo   = lane & 15;
  const int hi   = lane >> 4;

  v8f accum[2][2];
#pragma unroll
  for (int tm = 0; tm < 2; ++tm)
#pragma unroll
    for (int tn = 0; tn < 2; ++tn)
      accum[tm][tn] = v8f{0.f, 0.f, 0.f, 0.f, 0.f, 0.f, 0.f, 0.f};

  prefetch(0);
  commit(0);
  __syncthreads();

  for (int kt = 0; kt < KSTEPSF; ++kt) {
    const int buf = kt & 1;
    if (kt + 1 < KSTEPSF) prefetch(kt + 1);

    const unsigned short* A  = sA[buf];
    const unsigned short* Bs = sB[buf];
#pragma unroll
    for (int tm = 0; tm < 2; ++tm) {
      const unsigned short* ap = A + (msub + tm * 16 + lo) * LDTF;
      const v16bf af = load_frag(ap + hi * 8, ap + 16 + hi * 8);
#pragma unroll
      for (int tn = 0; tn < 2; ++tn) {
        const unsigned short* bpp = Bs + (nsub + tn * 16 + lo) * LDTF + hi * 8;
        const v16bf bfv = load_frag(bpp, bpp + 16);
        accum[tm][tn] = __builtin_amdgcn_wmma_f32_16x16x32_bf16(
            false, af, false, bfv, (short)0, accum[tm][tn], false, false);
      }
    }

    if (kt + 1 < KSTEPSF) {
      commit((kt + 1) & 1);
      __syncthreads();
    }
  }

#pragma unroll
  for (int tm = 0; tm < 2; ++tm) {
#pragma unroll
    for (int tn = 0; tn < 2; ++tn) {
      const int p  = p0 + nsub + tn * 16 + lo;
      const int n  = p / HWSZ;
      const int hw = p - n * HWSZ;
#pragma unroll
      for (int r = 0; r < 8; ++r) {
        const int co = co0 + msub + tm * 16 + r + hi * 8;
        Out[((size_t)n * COUT + co) * HWSZ + hw] = accum[tm][tn][r] + Bias[co];
      }
    }
  }
}

extern "C" void kernel_launch(void* const* d_in, const int* in_sizes, int n_in,
                              void* d_out, int out_size, void* d_ws, size_t ws_size,
                              hipStream_t stream) {
  const float* x    = (const float*)d_in[0];
  const float* W    = (const float*)d_in[1];
  const float* bias = (const float*)d_in[2];
  float* out = (float*)d_out;
  (void)in_sizes; (void)n_in; (void)out_size;

  const size_t xbytes = (size_t)NIMG * CIN * HWSZ * 2;  // 25,690,112
  const size_t wbytes = (size_t)COUT * KDIM * 2;        //    589,824
  dim3 grid(NPIX / BN, COUT / BM);                      // (1568, 2)
  dim3 block(256);

  if (ws_size >= xbytes + wbytes) {
    unsigned short* Xb = (unsigned short*)d_ws;
    unsigned short* Wb = (unsigned short*)((char*)d_ws + xbytes);
    const int xquads = NIMG * CIN * HWSZ / 4;           // 3,211,264
    cvt_x_bf16<<<xquads / 256, 256, 0, stream>>>(x, Xb);
    cvt_w_bf16<<<(COUT * KDIM) / 256, 256, 0, stream>>>(W, Wb);
    conv3x3_wmma_bf16<<<grid, block, 0, stream>>>(Xb, Wb, bias, out);
  } else {
    conv3x3_fallback<<<grid, block, 0, stream>>>(x, W, bias, out);
  }
}
